// EmbeddedAtomPairsNeuralNetwork_3573412790708
// MI455X (gfx1250) — matrix operand order repaired and checked by reading, not compiled, not run on hardware
//
#include <hip/hip_runtime.h>

// ---------------- problem constants (match reference) ----------------
#define KDIM    128
#define HDIM    128
#define NBLK    5
#define NILAY   2
#define NOLAY   2
#define NOUT    2
#define NLAY    20           // NBLK * (NILAY + NOLAY)
#define NATOMS  10000
#define NPAIRS  250000
#define TILE    128
#define PITCH   136          // halves per LDS row (128 + 8 pad), keeps 16B align
#define NTHREADS 256
#define SRCUT   10.0f
#define WMAT    (HDIM * HDIM)            // elements per weight matrix
#define WS_W16_BYTE_OFF 409600           // f16 weights after blockOut in d_ws

typedef __attribute__((ext_vector_type(16))) _Float16     v16h;
typedef __attribute__((ext_vector_type(8)))  float        v8f;
typedef __attribute__((ext_vector_type(4)))  unsigned int v4u;
typedef __attribute__((ext_vector_type(8)))  int          v8i;
typedef __attribute__((ext_vector_type(4)))  int          v4i;

union HF16 { v16h h; struct { v4u lo, hi; } u; };

#if __has_builtin(__builtin_amdgcn_tensor_load_to_lds)
#define HAVE_TDM 1
#else
#define HAVE_TDM 0
#endif

// shifted softplus: log(1+exp(x)) - log(2), numerically stable
__device__ __forceinline__ float ssp(float x) {
    float ax = fabsf(x);
    return fmaxf(x, 0.0f) + __logf(1.0f + __expf(-ax)) - 0.69314718056f;
}

#if HAVE_TDM
// Issue one TDM DMA: 128x128 f16 tile, global (row stride 128 halves) -> LDS
// with padding 4 DWORDs after every 64 DWORDs => LDS pitch of 136 halves.
// D# layout per CDNA5 ISA ch.8 (group0: count/lds_addr/global_addr/type,
// group1: data_size, pad cfg, dims, strides).  Wave-level op, EXEC ignored.
// This toolchain's builtin takes 6 args (extra int32x8 before cpol).
__device__ __forceinline__ void tdm_issue(const _Float16* gsrc, unsigned ldsByte) {
    unsigned long long ga = (unsigned long long)(uintptr_t)gsrc;
    v4u g0 = { 1u,                                   // count=1, user descriptor
               ldsByte,                              // lds_addr (bytes)
               (unsigned)(ga & 0xFFFFFFFFull),       // global_addr[31:0]
               (unsigned)((ga >> 32) & 0x01FFFFFFu) | (2u << 30) };  // addr[56:32] | type=2
    v8i g1 = { (int)((1u << 16)    // data_size = 2 bytes
                   | (1u << 20)    // pad_enable
                   | (5u << 22)    // pad_interval: 64 DWORDs (= one 128-half row)
                   | (3u << 25)),  // pad_amount: 4 DWORDs (= 8 halves)
               (int)(128u << 16),  // tensor_dim0 = 128 (bits 79:48)
               (int)(128u << 16),  // tensor_dim1 = 128 (bits 111:80)
               (int)(128u << 16),  // tile_dim0   = 128 (bits 127:112)
               128,                // tile_dim1   = 128 (bits 143:128)
               128,                // tensor_dim0_stride = 128 (bits 207:160)
               (int)0x40000000u,   // tensor_dim1_stride = 16384 (bits 255:208)
               0 };
    v4i z4 = { 0, 0, 0, 0 };
    v8i z8 = { 0, 0, 0, 0, 0, 0, 0, 0 };
    __builtin_amdgcn_tensor_load_to_lds(g0, g1, z4, z4, z8, 0);
}
#endif

// Fallback / non-TDM path: plain vectorized f16 copy global -> LDS (pitch 136)
__device__ __forceinline__ void copy_w16(const _Float16* __restrict__ g,
                                         _Float16* __restrict__ l, int tid) {
#pragma unroll
    for (int it = 0; it < 8; ++it) {
        const int idx  = it * NTHREADS + tid;   // 2048 x 16B chunks
        const int row  = idx >> 4;              // 16 chunks per 128-half row
        const int colv = (idx & 15) << 3;
        *(v4u*)(l + row * PITCH + colv) = *(const v4u*)(g + row * 128 + colv);
    }
}

// ---------------------------------------------------------------------
// 128x128 tile GEMM: dst = ssp(src @ W + bias), all operands in LDS f16.
// 8 waves; wave w owns rows [16w,16w+16) x full N=128.
// ---------------------------------------------------------------------
__device__ __forceinline__ void tile_gemm(const _Float16* __restrict__ src,
                                          _Float16* __restrict__ dst,
                                          const _Float16* __restrict__ wsm,
                                          const float* __restrict__ bsm,
                                          int tid) {
    const int lane = tid & 31;
    const int m0   = (tid >> 5) << 4;
    const int rowA = m0 + (lane & 15);
    const int kgrp = (lane >> 4) << 3;

    HF16 a[4];
#pragma unroll
    for (int kc = 0; kc < 4; ++kc) {
        const _Float16* pa = src + rowA * PITCH + kc * 32 + kgrp;
        a[kc].u.lo = *(const v4u*)pa;
        a[kc].u.hi = *(const v4u*)(pa + 16);
    }

#pragma unroll
    for (int nt = 0; nt < 8; ++nt) {
        const int n0 = nt << 4;
        v8f acc = {};                       // zero C -> inline-0 WMMA operand
#pragma unroll
        for (int kc = 0; kc < 4; ++kc) {
            HF16 bf;
            const _Float16* pb = wsm + (kc * 32 + kgrp) * PITCH + n0 + (lane & 15);
#pragma unroll
            for (int s = 0; s < 16; ++s)
                bf.h[s] = pb[(((s & 8) << 1) + (s & 7)) * PITCH];
            acc = __builtin_amdgcn_wmma_f32_16x16x32_f16(
                      false, a[kc].h, false, bf.h, (short)0, acc, false, false);
        }
        const float bv   = bsm[n0 + (lane & 15)];
        const int   rowD = m0 + ((lane >> 4) << 3);
        const int   colD = n0 + (lane & 15);
#pragma unroll
        for (int c = 0; c < 8; ++c)
            dst[(rowD + c) * PITCH + colD] = (_Float16)ssp(acc[c] + bv);
    }
}

__global__ __launch_bounds__(NTHREADS)
void physnet_edge_kernel(const float* __restrict__ R,
                         const int* __restrict__ idx_i,
                         const int* __restrict__ idx_j,
                         const _Float16* __restrict__ w16,      // 20 x 128x128 f16
                         const float* __restrict__ iB, const float* __restrict__ oB,
                         const float* __restrict__ oWf, const float* __restrict__ oBf,
                         float* __restrict__ rij_out,
                         float* __restrict__ blockOut) {
    extern __shared__ char smem[];
    _Float16* bufA = (_Float16*)smem;               // activations ping
    _Float16* bufB = bufA + TILE * PITCH;           // activations pong
    _Float16* bufC = bufB + TILE * PITCH;           // output-MLP result
    _Float16* ws0  = bufC + TILE * PITCH;           // weight buffer 0
    _Float16* ws1  = ws0  + HDIM * PITCH;           // weight buffer 1
    _Float16* oWfh = ws1  + HDIM * PITCH;           // head weights, 128x16 f16

    __shared__ float bsm[HDIM];
    __shared__ float er_s[TILE], fc_s[TILE];
    __shared__ int   pid_s[TILE];
    __shared__ float oBfs[NOUT];

    const int tid = threadIdx.x;
    const int tileBase = blockIdx.x * TILE;

#if HAVE_TDM
    const unsigned ldsDynBase = __builtin_amdgcn_groupstaticsize();
    const unsigned wsByte0 = ldsDynBase + (unsigned)(3 * TILE * PITCH) * 2u;
    const unsigned wsByte1 = wsByte0 + (unsigned)(HDIM * PITCH) * 2u;
    if (tid < 32) tdm_issue(w16, wsByte0);          // stage layer 0
#else
    copy_w16(w16, ws0, tid);
#endif

    // ---- per-pair distance + cutoff ----
    if (tid < TILE) {
        const int gp = tileBase + tid;
        float er = 0.f, fc = 0.f; int ii = -1;
        if (gp < NPAIRS) {
            const int i = idx_i[gp], j = idx_j[gp];
            const float dx = R[3*j+0] - R[3*i+0];
            const float dy = R[3*j+1] - R[3*i+1];
            const float dz = R[3*j+2] - R[3*i+2];
            const float r = sqrtf(dx*dx + dy*dy + dz*dz + 1e-12f);
            rij_out[gp] = r;
            const float xr = r * (1.0f / SRCUT);
            const float x2 = xr * xr;
            fc = (r < SRCUT) ? (1.0f - 6.f*x2*x2*xr + 15.f*x2*x2 - 10.f*x2*xr) : 0.f;
            er = __expf(-r);
            ii = i;
        }
        er_s[tid] = er; fc_s[tid] = fc; pid_s[tid] = ii;
    }
    __syncthreads();

    // ---- Gaussian basis on exp(-r), f16 into bufA ----
    {
        const float A0   = 4.5399929762484854e-05f;     // exp(-10)
        const float dmu  = (1.0f - A0) / (float)(KDIM - 1);
        const float c0   = (2.0f / (float)KDIM) * (1.0f - A0);
        const float wfac = 1.0f / (c0 * c0);
        const int p  = tid >> 1;
        const int kb = (tid & 1) << 6;
        const float er = er_s[p], fc = fc_s[p];
#pragma unroll 8
        for (int k = 0; k < 64; ++k) {
            const int kk = kb + k;
            const float t = er - (A0 + (float)kk * dmu);
            bufA[p * PITCH + kk] = (_Float16)(fc * __expf(-wfac * t * t));
        }
    }
    __syncthreads();

    // ---- 5 blocks x 4 layers, weights double-buffered one layer ahead ----
#pragma unroll 1
    for (int b = 0; b < NBLK; ++b) {
#pragma unroll 1
        for (int sub = 0; sub < 4; ++sub) {
            const int lay = b * 4 + sub;
#if HAVE_TDM
            if (tid < 32) {
                if (lay < NLAY - 1) {
                    tdm_issue(w16 + (size_t)(lay + 1) * WMAT,
                              ((lay + 1) & 1) ? wsByte1 : wsByte0);
                    __builtin_amdgcn_s_wait_tensorcnt((short)1);  // layer `lay` arrived
                } else {
                    __builtin_amdgcn_s_wait_tensorcnt((short)0);
                }
            }
#else
            if (lay < NLAY - 1)
                copy_w16(w16 + (size_t)(lay + 1) * WMAT, (lay & 1) ? ws0 : ws1, tid);
#endif
            const float* bias = (sub < 2) ? (iB + (size_t)(b * NILAY + sub) * HDIM)
                                          : (oB + (size_t)(b * NOLAY + sub - 2) * HDIM);
            if (tid < HDIM) bsm[tid] = bias[tid];
            __syncthreads();

            const _Float16* wcur = (lay & 1) ? ws1 : ws0;
            if      (sub == 0) tile_gemm(bufA, bufB, wcur, bsm, tid);
            else if (sub == 1) tile_gemm(bufB, bufA, wcur, bsm, tid);
            else if (sub == 2) tile_gemm(bufA, bufB, wcur, bsm, tid);
            else               tile_gemm(bufB, bufC, wcur, bsm, tid);

            if (sub == 3) {
                // stage head weights (128 x 16, cols >= NOUT zero) while gemm state drains
#pragma unroll
                for (int it = 0; it < 8; ++it) {
                    const int idx = it * NTHREADS + tid;    // 0..2047
                    const int k = idx >> 4, o = idx & 15;
                    oWfh[idx] = (o < NOUT) ? (_Float16)oWf[(b * HDIM + k) * NOUT + o]
                                           : (_Float16)0.f;
                }
                if (tid < NOUT) oBfs[tid] = oBf[b * NOUT + tid];
            }
            __syncthreads();
        }

        // ---- linear head via WMMA: out_e = y @ oWf(pad16) + oBf ----
        {
            const int lane = tid & 31;
            const int m0   = (tid >> 5) << 4;
            const int rowA = m0 + (lane & 15);
            const int kgrp = (lane >> 4) << 3;
            v8f acc = {};
#pragma unroll
            for (int kc = 0; kc < 4; ++kc) {
                HF16 a, bf;
                const _Float16* pa = bufC + rowA * PITCH + kc * 32 + kgrp;
                a.u.lo = *(const v4u*)pa;
                a.u.hi = *(const v4u*)(pa + 16);
                const _Float16* pb = oWfh + (kc * 32 + kgrp) * 16 + (lane & 15);
#pragma unroll
                for (int s = 0; s < 16; ++s)
                    bf.h[s] = pb[(((s & 8) << 1) + (s & 7)) * 16];
                acc = __builtin_amdgcn_wmma_f32_16x16x32_f16(
                          false, a.h, false, bf.h, (short)0, acc, false, false);
            }
            const int col = lane & 15;
            if (col < NOUT) {                       // only cols 0,1 are real outputs
                const int rbase = m0 + ((lane >> 4) << 3);
                const float bo = oBfs[col];
#pragma unroll
                for (int c = 0; c < 8; ++c) {
                    const int ii = pid_s[rbase + c];
                    if (ii >= 0)
                        atomicAdd(&blockOut[(size_t)b * NATOMS * NOUT + ii * NOUT + col],
                                  acc[c] + bo);
                }
            }
        }
        __syncthreads();
    }
}

// One-time: convert 20 weight matrices f32->f16 into ws; zero accumulators.
__global__ void prep_kernel(const float* __restrict__ iW, const float* __restrict__ oW,
                            _Float16* __restrict__ w16,
                            float* __restrict__ blockOut, float* __restrict__ d_out) {
    const int t = blockIdx.x * blockDim.x + threadIdx.x;
    if (t < NLAY * WMAT) {
        const int lay = t >> 14;                 // / 16384
        const int r   = t & (WMAT - 1);
        const int b   = lay >> 2, sub = lay & 3;
        const float* src = (sub < 2) ? (iW + (size_t)(b * NILAY + sub) * WMAT)
                                     : (oW + (size_t)(b * NOLAY + sub - 2) * WMAT);
        w16[t] = (_Float16)src[r];
    }
    if (t < NBLK * NATOMS * NOUT) blockOut[t] = 0.f;
    if (t == 0) d_out[NATOMS * NOUT + NPAIRS] = 0.f;   // nhloss slot
}

__global__ void finalize_kernel(const float* __restrict__ blockOut,
                                const int* __restrict__ Z,
                                const float* __restrict__ scales,
                                const float* __restrict__ shifts,
                                float* __restrict__ d_out) {
    const int t = blockIdx.x * blockDim.x + threadIdx.x;
    if (t >= NATOMS * NOUT) return;
    const int n = t >> 1, o = t & 1;
    float prev2 = 0.f, total = 0.f, nh = 0.f;
#pragma unroll
    for (int b = 0; b < NBLK; ++b) {
        const float v = blockOut[(size_t)b * NATOMS * NOUT + t];
        total += v;
        const float v2 = v * v;
        if (b > 0) nh += v2 / (v2 + prev2 + 1e-7f);
        prev2 = v2;
    }
    const int z = Z[n];
    d_out[t] = total * scales[o * 95 + z] + shifts[o * 95 + z];
    atomicAdd(&d_out[NATOMS * NOUT + NPAIRS], nh * (1.0f / (float)(NATOMS * NOUT)));
}

extern "C" void kernel_launch(void* const* d_in, const int* in_sizes, int n_in,
                              void* d_out, int out_size, void* d_ws, size_t ws_size,
                              hipStream_t stream) {
    (void)in_sizes; (void)n_in; (void)out_size; (void)ws_size;
    const int*   Z      = (const int*)  d_in[0];
    const float* R      = (const float*)d_in[1];
    const int*   idx_i  = (const int*)  d_in[2];
    const int*   idx_j  = (const int*)  d_in[3];
    const float* iW     = (const float*)d_in[4];
    const float* iB     = (const float*)d_in[5];
    const float* oW     = (const float*)d_in[6];
    const float* oB     = (const float*)d_in[7];
    const float* oWf    = (const float*)d_in[8];
    const float* oBf    = (const float*)d_in[9];
    const float* scales = (const float*)d_in[10];
    const float* shifts = (const float*)d_in[11];
    float* out = (float*)d_out;

    float*     blockOut = (float*)d_ws;                          // 400 KB
    _Float16*  w16      = (_Float16*)((char*)d_ws + WS_W16_BYTE_OFF);  // 640 KB

    prep_kernel<<<(NLAY * WMAT + 255) / 256, 256, 0, stream>>>(iW, oW, w16, blockOut, out);

    const size_t shmem = (size_t)(3 * TILE * PITCH + 2 * HDIM * PITCH + HDIM * 16)
                         * sizeof(_Float16);
    const int ntiles = (NPAIRS + TILE - 1) / TILE;
    physnet_edge_kernel<<<ntiles, NTHREADS, shmem, stream>>>(
        R, idx_i, idx_j, w16, iB, oB, oWf, oBf,
        out + NATOMS * NOUT, blockOut);

    finalize_kernel<<<(NATOMS * NOUT + 255) / 256, 256, 0, stream>>>(
        blockOut, Z, scales, shifts, out);
}
